// GAT_56547539419748
// MI455X (gfx1250) — matrix operand order repaired and checked by reading, not compiled, not run on hardware
//
#include <hip/hip_runtime.h>
#include <hip/hip_bf16.h>

typedef __bf16 bf16;
typedef bf16  v16bf __attribute__((ext_vector_type(16)));
typedef float v8f   __attribute__((ext_vector_type(8)));

#define GAT_ALPHA 0.1f
#define GAT_NEGINF (-9.0e15f)

#define BB 8
#define NNODE 2048
#define IND 512
#define OUTD 256
#define NHEAD 3
#define MROWS (BB * NNODE)   // 16384

// ---------------- workspace layout (bytes, all 256-aligned) ----------------
#define OFF_EMBB   ((size_t)0)                       // 16384*512 bf16
#define OFF_WTH    ((size_t)16777216)                // 3*256*512 bf16 (W_heads^T)
#define OFF_WOUTT  ((size_t)17563648)                // 256*768 bf16  (W_out^T)
#define OFF_WHT    ((size_t)17956864)                // 3*8*256*2048 bf16 (Wh^T)
#define OFF_X      ((size_t)43122688)                // 16384*768 bf16 (concat ELU)
#define OFF_F1     ((size_t)68288512)                // 3*16384 f32
#define OFF_F2     ((size_t)68485120)
#define OFF_MR     ((size_t)68681728)
#define OFF_SR     ((size_t)68878336)
#define WS_NEEDED  ((size_t)69074944)

// ---------------- prep kernels ----------------
__global__ void k_f32_to_bf16(const float* __restrict__ in, bf16* __restrict__ out, int n4) {
    int i = blockIdx.x * blockDim.x + threadIdx.x;
    if (i < n4) {
        float4 v = ((const float4*)in)[i];
        union { bf16 e[4]; uint2 u; } pk;
        pk.e[0] = (bf16)v.x; pk.e[1] = (bf16)v.y; pk.e[2] = (bf16)v.z; pk.e[3] = (bf16)v.w;
        *((uint2*)(out + (size_t)i * 4)) = pk.u;
    }
}

// in: [head][R][C] f32  ->  out: [head][C][R] bf16
__global__ void k_transpose_bf16(const float* __restrict__ in, bf16* __restrict__ out, int R, int C) {
    int total = R * C;
    int idx = blockIdx.x * 256 + threadIdx.x;
    size_t h = (size_t)blockIdx.y * (size_t)total;
    if (idx < total) {
        int k = idx % R;
        int n = idx / R;
        out[h + (size_t)n * R + k] = (bf16)in[h + (size_t)k * C + n];
    }
}

// ---------------- GEMM: C = A(row-major bf16, MxK) * B (given as Bt[n][k] bf16) ----------------
// block: 256 thr = 8 waves; block tile 32 rows x 256 cols; wave tile 16x64.
// epilogue: store C transposed bf16: Ct[head][b][n][i], i in [0,2048)
__global__ __launch_bounds__(256) void
k_gemm_tn(const bf16* __restrict__ A, int lda,
          const bf16* __restrict__ Bt, int ldb, size_t b_head_stride,
          bf16* __restrict__ Ct, size_t c_head_stride, int K) {
    const int tid = threadIdx.x;
    const int wave = tid >> 5, lane = tid & 31;
    const int rg = wave >> 2, cg = wave & 3;
    const int lanelo = lane & 15, hi = lane >> 4;
    const int row0 = blockIdx.x * 32 + rg * 16;
    const int col0 = cg * 64;
    const bf16* bt = Bt + (size_t)blockIdx.y * b_head_stride;

    v8f acc0, acc1, acc2, acc3;
    for (int r = 0; r < 8; ++r) { acc0[r] = 0.f; acc1[r] = 0.f; acc2[r] = 0.f; acc3[r] = 0.f; }

    const bf16* arow = A + (size_t)(row0 + lanelo) * lda + hi * 8;
    for (int k0 = 0; k0 < K; k0 += 32) {
        union { v16bf v; float4 q[2]; } fa;
        fa.q[0] = *(const float4*)(arow + k0);
        fa.q[1] = *(const float4*)(arow + k0 + 16);
#define GEMM_STEP(F, ACC) { \
        const bf16* bp = bt + (size_t)(col0 + F * 16 + lanelo) * ldb + k0 + hi * 8; \
        union { v16bf v; float4 q[2]; } fb; \
        fb.q[0] = *(const float4*)bp; fb.q[1] = *(const float4*)(bp + 16); \
        ACC = __builtin_amdgcn_wmma_f32_16x16x32_bf16(false, fa.v, false, fb.v, (short)0, ACC, false, false); }
        GEMM_STEP(0, acc0)
        GEMM_STEP(1, acc1)
        GEMM_STEP(2, acc2)
        GEMM_STEP(3, acc3)
#undef GEMM_STEP
    }

    const int b = row0 >> 11;
    const int ibase = (row0 & 2047) + hi * 8;
    bf16* cth = Ct + (size_t)blockIdx.y * c_head_stride + ((size_t)b * OUTD) * NNODE;
#define GEMM_ST(F, ACC) { \
    int n = col0 + F * 16 + lanelo; \
    union { bf16 e[8]; float4 q; } pk; \
    for (int r = 0; r < 8; ++r) pk.e[r] = (bf16)ACC[r]; \
    *(float4*)(cth + (size_t)n * NNODE + ibase) = pk.q; }
    GEMM_ST(0, acc0)
    GEMM_ST(1, acc1)
    GEMM_ST(2, acc2)
    GEMM_ST(3, acc3)
#undef GEMM_ST
}

// ---------------- f1/f2: per-row dot of Wh with attention vector halves ----------------
__global__ __launch_bounds__(256) void
k_f1f2(const bf16* __restrict__ whT, size_t head_stride,
       const float* __restrict__ avec,
       float* __restrict__ F1, float* __restrict__ F2) {
    const int h = blockIdx.y;
    const int m = blockIdx.x * 256 + threadIdx.x;
    const int b = m >> 11, i = m & 2047;
    const bf16* base = whT + (size_t)h * head_stride + ((size_t)b * OUTD) * NNODE + i;
    const float* a1 = avec + (size_t)h * 2 * OUTD;
    const float* a2 = a1 + OUTD;
    float f1 = 0.f, f2 = 0.f;
    for (int n = 0; n < OUTD; ++n) {
        float v = (float)base[(size_t)n * NNODE];
        f1 += v * a1[n];
        f2 += v * a2[n];
    }
    F1[(size_t)h * MROWS + m] = f1;
    F2[(size_t)h * MROWS + m] = f2;
}

// ---------------- rowstats: per-row online masked softmax (max, sumexp) ----------------
__global__ __launch_bounds__(256) void
k_rowstats(const float* __restrict__ adj, const float* __restrict__ F1, const float* __restrict__ F2,
           float* __restrict__ MR, float* __restrict__ SR) {
    const int h = blockIdx.y;
    const int wave = threadIdx.x >> 5, lane = threadIdx.x & 31;
    const int m = blockIdx.x * 8 + wave;
    const int b = m >> 11, i = m & 2047;
    const size_t ho = (size_t)h * MROWS;
    const float f1 = F1[ho + m];
    const float* arow = adj + ((size_t)b * NNODE + i) * NNODE;
    const float* f2p = F2 + ho + (size_t)b * NNODE;
    float mx = -3.0e38f, s = 0.f;
    for (int t = 0; t < NNODE / 32; ++t) {
        int j = t * 32 + lane;
        float e = f1 + f2p[j];
        e = (e > 0.f) ? e : GAT_ALPHA * e;
        e = (arow[j] > 0.f) ? e : GAT_NEGINF;
        float mn = fmaxf(mx, e);
        s = s * __expf(mx - mn) + __expf(e - mn);
        mx = mn;
    }
    for (int d = 16; d > 0; d >>= 1) {
        float mo = __shfl_xor(mx, d, 32);
        float so = __shfl_xor(s, d, 32);
        float mn = fmaxf(mx, mo);
        s = s * __expf(mx - mn) + so * __expf(mo - mn);
        mx = mn;
    }
    if (lane == 0) { MR[ho + m] = mx; SR[ho + m] = s; }
}

// ---------------- attention: out = softmax(masked e) @ Wh (flash-free: stats precomputed) ----------------
// block: 256 thr, 32 query rows x 256 cols. att tiles staged in LDS as WMMA A fragments.
__global__ __launch_bounds__(256) void
k_attn(const float* __restrict__ adj, const bf16* __restrict__ whT, size_t wh_head_stride,
       const float* __restrict__ F1, const float* __restrict__ F2,
       const float* __restrict__ MR, const float* __restrict__ SR,
       bf16* __restrict__ outb, float* __restrict__ outf,
       int ldo, int head_col_stride, int do_elu) {
    __shared__ __align__(16) bf16 attA[2][16][40];   // padded: conflict-free b128 reads
    __shared__ float eps[8][16][64];                 // per-wave epilogue transpose

    const int h = blockIdx.y;
    const int tid = threadIdx.x;
    const int row0 = blockIdx.x * 32;
    const int b = row0 >> 11;
    const int i0 = row0 & 2047;
    const size_t ho = (size_t)h * MROWS;

    // producer mapping: 4 att values per thread, fixed (rowgroup,row,jbase)
    const int v0 = tid * 4;
    const int jl0 = v0 & 31;
    const int mloc = (v0 >> 5) & 15;
    const int prg = v0 >> 9;
    const int mg = row0 + prg * 16 + mloc;
    const float f1v = F1[ho + mg];
    const float mv = MR[ho + mg];
    const float isv = 1.f / SR[ho + mg];
    const float* arow = adj + ((size_t)b * NNODE + (i0 + prg * 16 + mloc)) * NNODE + jl0;
    const float* f2p = F2 + ho + (size_t)b * NNODE + jl0;

    // consumer mapping
    const int wave = tid >> 5, lane = tid & 31;
    const int rg = wave >> 2, cg = wave & 3;
    const int lanelo = lane & 15, hi = lane >> 4;
    const bf16* wb = whT + (size_t)h * wh_head_stride + ((size_t)b * OUTD) * NNODE;

    v8f acc0, acc1, acc2, acc3;
    for (int r = 0; r < 8; ++r) { acc0[r] = 0.f; acc1[r] = 0.f; acc2[r] = 0.f; acc3[r] = 0.f; }

    for (int j0 = 0; j0 < NNODE; j0 += 32) {
        float4 av = *(const float4*)(arow + j0);
        float4 fv = *(const float4*)(f2p + j0);
        float aj[4] = { av.x, av.y, av.z, av.w };
        float f2v[4] = { fv.x, fv.y, fv.z, fv.w };
#pragma unroll
        for (int t = 0; t < 4; ++t) {
            float e = f1v + f2v[t];
            e = (e > 0.f) ? e : GAT_ALPHA * e;
            e = (aj[t] > 0.f) ? e : GAT_NEGINF;
            attA[prg][mloc][jl0 + t] = (bf16)(__expf(e - mv) * isv);
        }
        __syncthreads();

        union { v16bf v; float4 q[2]; } fa;
        fa.q[0] = *(const float4*)&attA[rg][lanelo][hi * 8];
        fa.q[1] = *(const float4*)&attA[rg][lanelo][16 + hi * 8];
#define ATT_STEP(F, ACC) { \
        const bf16* bp = wb + (size_t)(cg * 64 + F * 16 + lanelo) * NNODE + j0 + hi * 8; \
        union { v16bf v; float4 q[2]; } fb; \
        fb.q[0] = *(const float4*)bp; fb.q[1] = *(const float4*)(bp + 16); \
        ACC = __builtin_amdgcn_wmma_f32_16x16x32_bf16(false, fa.v, false, fb.v, (short)0, ACC, false, false); }
        ATT_STEP(0, acc0)
        ATT_STEP(1, acc1)
        ATT_STEP(2, acc2)
        ATT_STEP(3, acc3)
#undef ATT_STEP
        __syncthreads();
    }

    // epilogue: transpose 16x64 wave tile through LDS (per-wave region; DS in-order within wave)
#pragma unroll
    for (int r = 0; r < 8; ++r) {
        eps[wave][r + hi * 8][lanelo]      = acc0[r];
        eps[wave][r + hi * 8][16 + lanelo] = acc1[r];
        eps[wave][r + hi * 8][32 + lanelo] = acc2[r];
        eps[wave][r + hi * 8][48 + lanelo] = acc3[r];
    }
    const int orow = lanelo;
    const int mgo = row0 + rg * 16 + orow;
    if (outb) {
        bf16* op = outb + (size_t)mgo * ldo + h * head_col_stride + cg * 64 + hi * 32;
#pragma unroll
        for (int c8 = 0; c8 < 4; ++c8) {
            union { bf16 e[8]; float4 q; } pk;
#pragma unroll
            for (int c = 0; c < 8; ++c) {
                float v = eps[wave][orow][hi * 32 + c8 * 8 + c];
                if (do_elu) v = (v > 0.f) ? v : (__expf(v) - 1.f);
                pk.e[c] = (bf16)v;
            }
            *(float4*)(op + c8 * 8) = pk.q;
        }
    } else {
        float* op = outf + (size_t)mgo * ldo + cg * 64 + hi * 32;
#pragma unroll
        for (int c4 = 0; c4 < 8; ++c4) {
            float4 q;
            q.x = eps[wave][orow][hi * 32 + c4 * 4 + 0];
            q.y = eps[wave][orow][hi * 32 + c4 * 4 + 1];
            q.z = eps[wave][orow][hi * 32 + c4 * 4 + 2];
            q.w = eps[wave][orow][hi * 32 + c4 * 4 + 3];
            *(float4*)(op + c4 * 4) = q;
        }
    }
}

// ---------------- host launch ----------------
extern "C" void kernel_launch(void* const* d_in, const int* in_sizes, int n_in,
                              void* d_out, int out_size, void* d_ws, size_t ws_size,
                              hipStream_t stream) {
    (void)in_sizes; (void)n_in; (void)out_size;
    if (ws_size < WS_NEEDED) return;

    const float* emb = (const float*)d_in[0];   // [8,2048,512]
    const float* adj = (const float*)d_in[1];   // [8,2048,2048]
    const float* Wh  = (const float*)d_in[2];   // [3,512,256]
    const float* ah  = (const float*)d_in[3];   // [3,512,1]
    const float* Wo  = (const float*)d_in[4];   // [768,256]
    const float* ao  = (const float*)d_in[5];   // [512,1]
    float* out = (float*)d_out;                 // [8,2048,256]

    char* ws = (char*)d_ws;
    bf16* embb  = (bf16*)(ws + OFF_EMBB);
    bf16* WtH   = (bf16*)(ws + OFF_WTH);
    bf16* WoutT = (bf16*)(ws + OFF_WOUTT);
    bf16* WhT   = (bf16*)(ws + OFF_WHT);
    bf16* X     = (bf16*)(ws + OFF_X);
    float* F1 = (float*)(ws + OFF_F1);
    float* F2 = (float*)(ws + OFF_F2);
    float* MR = (float*)(ws + OFF_MR);
    float* SR = (float*)(ws + OFF_SR);

    const size_t whHeadStride = (size_t)BB * OUTD * NNODE;   // elems
    const int LDX = NHEAD * OUTD;                            // 768

    // prep
    k_f32_to_bf16<<<dim3((MROWS * IND / 4 + 255) / 256), 256, 0, stream>>>(emb, embb, MROWS * IND / 4);
    k_transpose_bf16<<<dim3((IND * OUTD + 255) / 256, NHEAD), 256, 0, stream>>>(Wh, WtH, IND, OUTD);
    k_transpose_bf16<<<dim3((LDX * OUTD + 255) / 256, 1), 256, 0, stream>>>(Wo, WoutT, LDX, OUTD);

    // ---- layer 1 (3 heads) ----
    k_gemm_tn<<<dim3(MROWS / 32, NHEAD), 256, 0, stream>>>(
        embb, IND, WtH, IND, (size_t)OUTD * IND, WhT, whHeadStride, IND);
    k_f1f2<<<dim3(MROWS / 256, NHEAD), 256, 0, stream>>>(WhT, whHeadStride, ah, F1, F2);
    k_rowstats<<<dim3(MROWS / 8, NHEAD), 256, 0, stream>>>(adj, F1, F2, MR, SR);
    k_attn<<<dim3(MROWS / 32, NHEAD), 256, 0, stream>>>(
        adj, WhT, whHeadStride, F1, F2, MR, SR, X, nullptr, LDX, OUTD, 1);

    // ---- layer 2 (single head, reuse head-0 scratch slots) ----
    k_gemm_tn<<<dim3(MROWS / 32, 1), 256, 0, stream>>>(
        X, LDX, WoutT, LDX, 0, WhT, 0, LDX);
    k_f1f2<<<dim3(MROWS / 256, 1), 256, 0, stream>>>(WhT, 0, ao, F1, F2);
    k_rowstats<<<dim3(MROWS / 8, 1), 256, 0, stream>>>(adj, F1, F2, MR, SR);
    k_attn<<<dim3(MROWS / 32, 1), 256, 0, stream>>>(
        adj, WhT, 0, F1, F2, MR, SR, nullptr, out, OUTD, 0, 0);
}
